// Model_39444979646959
// MI455X (gfx1250) — compile-verified, matrix operand order checked by reading, not run on hardware
//
#include <hip/hip_runtime.h>
#include <hip/hip_bf16.h>

typedef unsigned short u16;
typedef __attribute__((ext_vector_type(16))) __bf16 v16bf;
typedef __attribute__((ext_vector_type(8)))  float  v8f;

union Frag16 {
    uint4  q[2];
    u16    h[16];
    v16bf  v;
};

// ---- gfx1250 async global->LDS path (ASYNCcnt tracked), with safe fallback ----
#if defined(__has_builtin)
#  if __has_builtin(__builtin_amdgcn_global_load_async_to_lds_b128)
#    define HAVE_ASYNC_LDS 1
#  endif
#  if __has_builtin(__builtin_amdgcn_s_wait_asynccnt)
#    define HAVE_WAIT_ASYNC 1
#  endif
#endif
#ifndef HAVE_ASYNC_LDS
#  define HAVE_ASYNC_LDS 0
#endif
#ifndef HAVE_WAIT_ASYNC
#  define HAVE_WAIT_ASYNC 0
#endif

#if HAVE_ASYNC_LDS
typedef int i4v __attribute__((vector_size(16)));
typedef __attribute__((address_space(1))) i4v* gas_i4p;   // global int4*
typedef __attribute__((address_space(3))) i4v* las_i4p;   // LDS int4*
#endif

__device__ __forceinline__ void stage_b128(const u16* g, u16* l) {
#if HAVE_ASYNC_LDS
    __builtin_amdgcn_global_load_async_to_lds_b128((gas_i4p)g, (las_i4p)l, 0, 0);
#else
    *(uint4*)l = *(const uint4*)g;
#endif
}

template <int N>
__device__ __forceinline__ void wait_async() {
#if HAVE_ASYNC_LDS
#  if HAVE_WAIT_ASYNC
    __builtin_amdgcn_s_wait_asynccnt(N);
#  else
    asm volatile("s_wait_asynccnt %0" :: "i"(N) : "memory");
#  endif
#endif
}

__device__ __forceinline__ u16 f2bf(float f) {
    unsigned int u = __float_as_uint(f);
    unsigned int r = u + 0x7FFFu + ((u >> 16) & 1u);   // round-to-nearest-even
    return (u16)(r >> 16);
}

// ---------------------------------------------------------------------------
// fp32 -> bf16 elementwise conversion
// ---------------------------------------------------------------------------
__global__ void cvt_f32_bf16(const float* __restrict__ in, u16* __restrict__ out, int n) {
    int i = blockIdx.x * blockDim.x + threadIdx.x;
    if (i < n) out[i] = f2bf(in[i]);
}

// conv weight repack: (Cout, Cin, 3) fp32 -> [k][Cout][Cin] bf16
__global__ void repack_convw(const float* __restrict__ in, u16* __restrict__ out,
                             int Cout, int Cin) {
    int i = blockIdx.x * blockDim.x + threadIdx.x;
    int n = Cout * Cin * 3;
    if (i >= n) return;
    int co  = i / (Cin * 3);
    int rem = i - co * (Cin * 3);
    int ci  = rem / 3;
    int k   = rem - ci * 3;
    out[(size_t)k * Cout * Cin + (size_t)co * Cin + ci] = f2bf(in[i]);
}

// ---------------------------------------------------------------------------
// bf16 GEMM:  C(MxN) = A(MxK) * W(NxK)^T + bias   (torch Linear layout)
// 4-wave block computes a 64x64 tile. The shared 64-col B strip is staged
// into LDS (async global->LDS, double buffered); each wave owns one 16-row
// M-tile and 4 accumulators. LDS row stride padded to 40 u16 (80B) to spread
// ds_load_b128 row addresses across bank groups.
// ---------------------------------------------------------------------------
#define LDSB_STRIDE 40
__global__ __launch_bounds__(128)
void gemm_bf16(const u16* __restrict__ A, const u16* __restrict__ Wt,
               const float* __restrict__ bias, void* __restrict__ out,
               int M, int N, int K, int out_bf16) {
    __shared__ u16 ldsB[2][64 * LDSB_STRIDE];   // [buf][row(n) * 40 + k]

    const int tid   = threadIdx.x;
    const int lane  = tid & 31;
    const int wave  = tid >> 5;
    const int half  = lane >> 4;       // 0 or 1
    const int lm    = lane & 15;
    const int kbase = half * 8;

    const int m0 = blockIdx.y * 64 + wave * 16;
    const int n0 = blockIdx.x * 64;

    // staging: thread handles row = tid>>1, 32 contiguous bytes = two b128
    const int srow = tid >> 1;
    const int ssub = (tid & 1) * 16;           // u16 offset within the 32-u16 row
    const u16* gstage = Wt + (size_t)(n0 + srow) * K + ssub;
    u16*       lstage0 = &ldsB[0][srow * LDSB_STRIDE + ssub];
    u16*       lstage1 = &ldsB[1][srow * LDSB_STRIDE + ssub];

    const u16* arow = A + (size_t)(m0 + lm) * K + kbase;

    v8f acc0 = {}, acc1 = {}, acc2 = {}, acc3 = {};

    // prologue: stage k0 = 0 into buf 0
    stage_b128(gstage, lstage0);
    stage_b128(gstage + 8, lstage0 + 8);

    int buf = 0;
    for (int k0 = 0; k0 < K; k0 += 32) {
        if (k0 + 32 < K) {   // stage next K-slice into the other buffer
            u16* ln = buf ? lstage0 : lstage1;
            const u16* gn = gstage + (k0 + 32);
            stage_b128(gn, ln);
            stage_b128(gn + 8, ln + 8);
            wait_async<2>();             // current buffer's loads complete
        } else {
            wait_async<0>();
        }
        __syncthreads();                 // staged data visible to all waves

        Frag16 fa, f0, f1, f2, f3;
        fa.q[0] = *(const uint4*)(arow + k0);
        fa.q[1] = *(const uint4*)(arow + k0 + 16);

        const u16* lb = ldsB[buf];
        f0.q[0] = *(const uint4*)(lb + (lm +  0) * LDSB_STRIDE + kbase);
        f0.q[1] = *(const uint4*)(lb + (lm +  0) * LDSB_STRIDE + kbase + 16);
        f1.q[0] = *(const uint4*)(lb + (lm + 16) * LDSB_STRIDE + kbase);
        f1.q[1] = *(const uint4*)(lb + (lm + 16) * LDSB_STRIDE + kbase + 16);
        f2.q[0] = *(const uint4*)(lb + (lm + 32) * LDSB_STRIDE + kbase);
        f2.q[1] = *(const uint4*)(lb + (lm + 32) * LDSB_STRIDE + kbase + 16);
        f3.q[0] = *(const uint4*)(lb + (lm + 48) * LDSB_STRIDE + kbase);
        f3.q[1] = *(const uint4*)(lb + (lm + 48) * LDSB_STRIDE + kbase + 16);

        acc0 = __builtin_amdgcn_wmma_f32_16x16x32_bf16(false, fa.v, false, f0.v, (short)0, acc0, false, false);
        acc1 = __builtin_amdgcn_wmma_f32_16x16x32_bf16(false, fa.v, false, f1.v, (short)0, acc1, false, false);
        acc2 = __builtin_amdgcn_wmma_f32_16x16x32_bf16(false, fa.v, false, f2.v, (short)0, acc2, false, false);
        acc3 = __builtin_amdgcn_wmma_f32_16x16x32_bf16(false, fa.v, false, f3.v, (short)0, acc3, false, false);

        __syncthreads();                 // protect buffer being re-staged next iter
        buf ^= 1;
    }

    #pragma unroll
    for (int t = 0; t < 4; ++t) {
        v8f acc = (t == 0) ? acc0 : (t == 1) ? acc1 : (t == 2) ? acc2 : acc3;
        int n = n0 + t * 16 + lm;
        float bv = bias[n];
        #pragma unroll
        for (int r = 0; r < 8; ++r) {
            int mr = m0 + r + half * 8;
            float v = acc[r] + bv;
            size_t idx = (size_t)mr * N + n;
            if (out_bf16) ((u16*)out)[idx] = f2bf(v);
            else          ((float*)out)[idx] = v;
        }
    }
}

// ---------------------------------------------------------------------------
// conv1d (k=3, pad=0, dilation d) as 3 accumulated GEMMs through WMMA.
// X: (B, Cin, L) bf16 ; WT: [k][Cout][Cin] bf16 ; Y: (B, Cout, Lout)
// grid: (ceil(Lout/16), Cout/16, B), one wave per block.
// ---------------------------------------------------------------------------
__global__ __launch_bounds__(32)
void conv1d_wmma(const u16* __restrict__ X, const u16* __restrict__ WT,
                 const float* __restrict__ bias, void* __restrict__ Y,
                 int Cin, int Cout, int L, int Lout, int dil, int out_bf16) {
    const int lane  = threadIdx.x;
    const int half  = lane >> 4;
    const int lm    = lane & 15;
    const int kbase = half * 8;

    const int b    = blockIdx.z;
    const int co_a = blockIdx.y * 16 + lm;      // A-fragment row (Cout)
    const int l_n  = blockIdx.x * 16 + lm;      // B-fragment column (Lout)
    const int l_ld = (l_n < Lout) ? l_n : (Lout - 1);  // clamp loads in pad cols

    const u16* xb = X + (size_t)b * Cin * L;
    v8f acc = {};

    for (int k = 0; k < 3; ++k) {
        const u16* wrow = WT + ((size_t)k * Cout + co_a) * Cin + kbase;
        const u16* xcol = xb + l_ld + k * dil;
        for (int ci0 = 0; ci0 < Cin; ci0 += 32) {
            Frag16 fa, fb;
            fa.q[0] = *(const uint4*)(wrow + ci0);
            fa.q[1] = *(const uint4*)(wrow + ci0 + 16);
            #pragma unroll
            for (int j = 0; j < 16; ++j) {
                int kk = kbase + ((j >> 3) << 4) + (j & 7);  // K index in 0..31
                fb.h[j] = xcol[(size_t)(ci0 + kk) * L];
            }
            acc = __builtin_amdgcn_wmma_f32_16x16x32_bf16(false, fa.v, false, fb.v, (short)0, acc, false, false);
        }
    }

    #pragma unroll
    for (int r = 0; r < 8; ++r) {
        int co = blockIdx.y * 16 + r + half * 8;
        float v = acc[r] + bias[co];
        if (l_n < Lout) {
            size_t idx = ((size_t)b * Cout + co) * Lout + l_n;
            if (out_bf16) ((u16*)Y)[idx] = f2bf(v);
            else          ((float*)Y)[idx] = v;
        }
    }
}

// ---------------------------------------------------------------------------
// Sinkhorn: out = (h*h + 0.001); 10x { row-normalize; col-normalize }.
// One 256-thread workgroup per sample; 64x64 tile resident in LDS (+pad col).
// ---------------------------------------------------------------------------
__global__ __launch_bounds__(256)
void sinkhorn64(const float* __restrict__ H, float* __restrict__ out) {
    __shared__ float mat[64][65];
    __shared__ float part[64][4];
    __shared__ float sums[64];

    const int t   = threadIdx.x;
    const int b   = blockIdx.x;
    const int row = t >> 2;          // 0..63
    const int cq  = (t & 3) * 16;    // col base: 0,16,32,48

    const float* hb = H + (size_t)b * 4096;
    #pragma unroll
    for (int i = 0; i < 16; ++i) {
        float v = hb[row * 64 + cq + i];
        mat[row][cq + i] = v * v + 0.001f;
    }
    __syncthreads();

    for (int it = 0; it < 10; ++it) {
        // ---- row normalize ----
        float s = 0.f;
        #pragma unroll
        for (int i = 0; i < 16; ++i) s += mat[row][cq + i];
        part[row][t & 3] = s;
        __syncthreads();
        if (t < 64) sums[t] = part[t][0] + part[t][1] + part[t][2] + part[t][3];
        __syncthreads();
        float rinv = 1.0f / sums[row];
        #pragma unroll
        for (int i = 0; i < 16; ++i) mat[row][cq + i] *= rinv;
        __syncthreads();

        // ---- col normalize ----  (thread t: column t>>2, rows cq..cq+15)
        s = 0.f;
        #pragma unroll
        for (int i = 0; i < 16; ++i) s += mat[cq + i][row];
        part[row][t & 3] = s;
        __syncthreads();
        if (t < 64) sums[t] = part[t][0] + part[t][1] + part[t][2] + part[t][3];
        __syncthreads();
        #pragma unroll
        for (int i = 0; i < 16; ++i) mat[row][cq + i] /= sums[cq + i];
        __syncthreads();
    }

    float* ob = out + (size_t)b * 4096;
    #pragma unroll
    for (int i = 0; i < 16; ++i) ob[row * 64 + cq + i] = mat[row][cq + i];
}

// ---------------------------------------------------------------------------
// Launcher
// ---------------------------------------------------------------------------
static const int Bm = 256;   // batch
static const int F  = 256;   // input features
static const int Nn = 64;    // model N

extern "C" void kernel_launch(void* const* d_in, const int* in_sizes, int n_in,
                              void* d_out, int out_size, void* d_ws, size_t ws_size,
                              hipStream_t stream) {
    const float* x   = (const float*)d_in[0];
    const float* w1  = (const float*)d_in[1];
    const float* b1  = (const float*)d_in[2];
    const float* w2  = (const float*)d_in[3];
    const float* b2  = (const float*)d_in[4];
    const float* w3  = (const float*)d_in[5];
    const float* b3  = (const float*)d_in[6];
    const float* cw1 = (const float*)d_in[7];   const float* cb1 = (const float*)d_in[8];
    const float* cw2 = (const float*)d_in[9];   const float* cb2 = (const float*)d_in[10];
    const float* cw3 = (const float*)d_in[11];  const float* cb3 = (const float*)d_in[12];
    const float* cw4 = (const float*)d_in[13];  const float* cb4 = (const float*)d_in[14];
    const float* cw5 = (const float*)d_in[15];  const float* cb5 = (const float*)d_in[16];

    // ---- workspace bump allocator (256B aligned) ----
    char* ws = (char*)d_ws;
    size_t off = 0;
    auto walloc = [&](size_t bytes) -> void* {
        void* p = ws + off;
        off += (bytes + 255) & ~(size_t)255;
        return p;
    };

    const int L1N = 2 * Nn;          // 128
    const int L2N = Nn * Nn;         // 4096
    const int L3N = 4 * Nn * Nn;     // 16384

    u16* xb   = (u16*)walloc((size_t)Bm * F * 2);
    u16* w1b  = (u16*)walloc((size_t)L1N * F * 2);
    u16* w2b  = (u16*)walloc((size_t)L2N * L1N * 2);
    u16* w3b  = (u16*)walloc((size_t)L3N * L2N * 2);
    u16* h1b  = (u16*)walloc((size_t)Bm * L1N * 2);
    u16* h2b  = (u16*)walloc((size_t)Bm * L2N * 2);
    u16* h3b  = (u16*)walloc((size_t)Bm * L3N * 2);
    u16* cw1b = (u16*)walloc((size_t)192 * 128 * 3 * 2);
    u16* cw2b = (u16*)walloc((size_t)128 * 192 * 3 * 2);
    u16* cw3b = (u16*)walloc((size_t)64 * 128 * 3 * 2);
    u16* cw4b = (u16*)walloc((size_t)64 * 64 * 3 * 2);
    u16* cw5b = (u16*)walloc((size_t)64 * 64 * 3 * 2);
    u16* a1   = (u16*)walloc((size_t)Bm * 192 * 72 * 2);
    u16* a2   = (u16*)walloc((size_t)Bm * 128 * 70 * 2);
    u16* a3   = (u16*)walloc((size_t)Bm * 64 * 68 * 2);
    u16* a4   = (u16*)walloc((size_t)Bm * 64 * 66 * 2);
    float* a5 = (float*)walloc((size_t)Bm * 64 * 64 * 4);

    auto cvt = [&](const float* src, u16* dst, int n) {
        cvt_f32_bf16<<<(n + 255) / 256, 256, 0, stream>>>(src, dst, n);
    };

    // ---- precision conversion / repack pass ----
    cvt(x,  xb,  Bm * F);
    cvt(w1, w1b, L1N * F);
    cvt(w2, w2b, L2N * L1N);
    cvt(w3, w3b, L3N * L2N);
    repack_convw<<<(192 * 128 * 3 + 255) / 256, 256, 0, stream>>>(cw1, cw1b, 192, 128);
    repack_convw<<<(128 * 192 * 3 + 255) / 256, 256, 0, stream>>>(cw2, cw2b, 128, 192);
    repack_convw<<<(64 * 128 * 3 + 255) / 256, 256, 0, stream>>>(cw3, cw3b, 64, 128);
    repack_convw<<<(64 * 64 * 3 + 255) / 256, 256, 0, stream>>>(cw4, cw4b, 64, 64);
    repack_convw<<<(64 * 64 * 3 + 255) / 256, 256, 0, stream>>>(cw5, cw5b, 64, 64);

    // ---- linear stack (WMMA GEMMs, 64x64 block tiles, async-LDS staged B) ----
    gemm_bf16<<<dim3(L1N / 64, Bm / 64), 128, 0, stream>>>(xb,  w1b, b1, h1b, Bm, L1N, F,   1);
    gemm_bf16<<<dim3(L2N / 64, Bm / 64), 128, 0, stream>>>(h1b, w2b, b2, h2b, Bm, L2N, L1N, 1);
    gemm_bf16<<<dim3(L3N / 64, Bm / 64), 128, 0, stream>>>(h2b, w3b, b3, h3b, Bm, L3N, L2N, 1);

    // h3b viewed as (B, 128, 128) bf16 — feeds conv stack directly.
    const int dil1 = (Nn - 8) / 2;  // 28
    conv1d_wmma<<<dim3(5, 192 / 16, Bm), 32, 0, stream>>>(h3b, cw1b, cb1, a1, 128, 192, 128, 72, dil1, 1);
    conv1d_wmma<<<dim3(5, 128 / 16, Bm), 32, 0, stream>>>(a1,  cw2b, cb2, a2, 192, 128, 72,  70, 1,    1);
    conv1d_wmma<<<dim3(5, 64 / 16,  Bm), 32, 0, stream>>>(a2,  cw3b, cb3, a3, 128, 64,  70,  68, 1,    1);
    conv1d_wmma<<<dim3(5, 64 / 16,  Bm), 32, 0, stream>>>(a3,  cw4b, cb4, a4, 64,  64,  68,  66, 1,    1);
    conv1d_wmma<<<dim3(4, 64 / 16,  Bm), 32, 0, stream>>>(a4,  cw5b, cb5, a5, 64,  64,  66,  64, 1,    0);

    // ---- square + sinkhorn into d_out ----
    sinkhorn64<<<Bm, 256, 0, stream>>>(a5, (float*)d_out);
}
